// MoELayer_84971632984718
// MI455X (gfx1250) — matrix operand order repaired and checked by reading, not compiled, not run on hardware
//
#include <hip/hip_runtime.h>
#include <math.h>

#define Bdim 4
#define Sdim 2048
#define Hdim 768
#define Idim 3072
#define Edim 8
#define NTOK (Bdim * Sdim) /* 8192 tokens */
#define MTILE 32           /* tokens per FFN block (2 x 16-row WMMA tiles) */
#define NB1 (Idim / 16)    /* 192 column blocks in W1 */
#define KB1 (Hdim / 32)    /* 24 K blocks in W1 */
#define NB2 (Hdim / 16)    /* 48 column blocks in W2 */
#define KB2 (Idim / 32)    /* 96 K blocks in W2 */

typedef __attribute__((ext_vector_type(16))) __bf16 v16bf;
typedef __attribute__((ext_vector_type(8))) float v8f;

// ---------------------------------------------------------------------------
// Kernel 1: zero output + ws counters (output accumulated with atomics).
// ---------------------------------------------------------------------------
__global__ void zero_kernel(float* __restrict__ out, size_t n_out,
                            int* __restrict__ counts,
                            float* __restrict__ importance) {
  size_t i = (size_t)blockIdx.x * blockDim.x + threadIdx.x;
  size_t stride = (size_t)gridDim.x * blockDim.x;
  for (size_t x = i; x < n_out; x += stride) out[x] = 0.f;
  if (blockIdx.x == 0 && threadIdx.x < Edim) {
    counts[threadIdx.x] = 0;
    importance[threadIdx.x] = 0.f;
  }
}

// ---------------------------------------------------------------------------
// Kernel 2: weight prep. Convert fp32 weights to bf16, pre-swizzled into the
// contiguous-k WMMA B layout used here, one 32B contiguous slice per (tile, lane):
//   tile = (e, nb, kb): K rows kb*32..+32, N cols nb*16..+16
//   lane l: column n=l&15, rows kb*32 + ((l&16)?16:0) + el, el=0..15
// Output index = ((e*NB + nb)*KB + kb)*32 + lane  (x16 bf16 elements).
// One thread per (tile, lane); gid ordering == output layout ordering.
// ---------------------------------------------------------------------------
__global__ __launch_bounds__(256) void prep_weights_kernel(
    const float* __restrict__ w1, const float* __restrict__ w2,
    __bf16* __restrict__ w1s, __bf16* __restrict__ w2s) {
  const int N1 = Edim * NB1 * KB1 * 32;  // 1179648 threads for W1
  int gid = blockIdx.x * 256 + threadIdx.x;
  if (gid < N1) {
    int lane = gid & 31;
    int t = gid >> 5;
    int kb = t % KB1; t /= KB1;
    int nb = t % NB1; int e = t / NB1;
    int n = lane & 15;
    int koff = (lane & 16) ? 16 : 0;
    const float* src =
        w1 + ((size_t)e * Hdim + kb * 32 + koff) * Idim + nb * 16 + n;
    v16bf v;
#pragma unroll
    for (int el = 0; el < 16; ++el) v[el] = (__bf16)src[(size_t)el * Idim];
    *(v16bf*)(w1s + (size_t)gid * 16) = v;
  } else {
    int g2 = gid - N1;
    int lane = g2 & 31;
    int t = g2 >> 5;
    int kb = t % KB2; t /= KB2;
    int nb = t % NB2; int e = t / NB2;
    int n = lane & 15;
    int koff = (lane & 16) ? 16 : 0;
    const float* src =
        w2 + ((size_t)e * Idim + kb * 32 + koff) * Hdim + nb * 16 + n;
    v16bf v;
#pragma unroll
    for (int el = 0; el < 16; ++el) v[el] = (__bf16)src[(size_t)el * Hdim];
    *(v16bf*)(w2s + (size_t)g2 * 16) = v;
  }
}

// ---------------------------------------------------------------------------
// Kernel 3: router. logits -> top2 -> softmax -> dense gating (to d_out),
// per-expert token lists + gate weights + importance sums (to ws).
// ---------------------------------------------------------------------------
__global__ __launch_bounds__(256) void router_kernel(
    const float* __restrict__ hs, const float* __restrict__ rw,
    const float* __restrict__ rb, float* __restrict__ gating,
    int* __restrict__ lists, float* __restrict__ gwbuf,
    int* __restrict__ counts, float* __restrict__ importance) {
  int t = blockIdx.x * blockDim.x + threadIdx.x;
  if (t >= NTOK) return;
  float logit[Edim];
#pragma unroll
  for (int e = 0; e < Edim; ++e) logit[e] = rb[e];
  const float* x = hs + (size_t)t * Hdim;
  for (int k = 0; k < Hdim; ++k) {
    float xv = x[k];
    const float* r = rw + k * Edim;
#pragma unroll
    for (int e = 0; e < Edim; ++e) logit[e] = fmaf(xv, r[e], logit[e]);
  }
  int i0 = 0;
  float v0 = logit[0];
#pragma unroll
  for (int e = 1; e < Edim; ++e)
    if (logit[e] > v0) { v0 = logit[e]; i0 = e; }
  int i1 = -1;
  float v1 = -INFINITY;
#pragma unroll
  for (int e = 0; e < Edim; ++e)
    if (e != i0 && logit[e] > v1) { v1 = logit[e]; i1 = e; }
  float ex = expf(v1 - v0);              // v0 >= v1, stable softmax over top-2
  float w0 = 1.f / (1.f + ex);
  float w1 = ex / (1.f + ex);
  gating[(size_t)t * Edim + i0] = w0;
  gating[(size_t)t * Edim + i1] = w1;
  int p0 = atomicAdd(&counts[i0], 1);
  lists[i0 * NTOK + p0] = t;
  gwbuf[i0 * NTOK + p0] = w0;
  int p1 = atomicAdd(&counts[i1], 1);
  lists[i1 * NTOK + p1] = t;
  gwbuf[i1 * NTOK + p1] = w1;
  atomicAdd(&importance[i0], w0);
  atomicAdd(&importance[i1], w1);
}

// ---------------------------------------------------------------------------
// Kernel 4: aux (load-balancing) loss. var(importance, ddof=1)/mean^2 * 0.01
// ---------------------------------------------------------------------------
__global__ void aux_kernel(const float* __restrict__ importance,
                           float* __restrict__ aux) {
  float mean = 0.f;
  for (int e = 0; e < Edim; ++e) mean += importance[e];
  mean /= (float)Edim;
  float var = 0.f;
  for (int e = 0; e < Edim; ++e) {
    float d = importance[e] - mean;
    var += d * d;
  }
  var /= (float)(Edim - 1);
  *aux = 0.01f * (var / (mean * mean + 1e-8f));
}

// ---------------------------------------------------------------------------
// Kernel 5: routed fused expert FFN with bf16 WMMA, M=32 tokens per block.
// grid = (NTOK/32 tiles, Edim experts), block = 256 threads (8 waves).
// B operands come pre-swizzled bf16 (2 coalesced global_load_b128 per tile),
// each A ds_load pair feeds 4 WMMAs (K-outer loop order).
// ---------------------------------------------------------------------------
__global__ __launch_bounds__(256) void moe_ffn_kernel(
    const float* __restrict__ hs, const __bf16* __restrict__ w1s,
    const float* __restrict__ b1, const __bf16* __restrict__ w2s,
    const float* __restrict__ b2, const int* __restrict__ lists,
    const float* __restrict__ gwbuf, const int* __restrict__ counts,
    float* __restrict__ out) {
  const int e = blockIdx.y;
  const int base = blockIdx.x * MTILE;
  const int cnt = counts[e];
  if (base >= cnt) return;  // uniform across block

  __shared__ __align__(32) __bf16 Xa[2][24][32][16];  // 48 KB: X, A-layout
  __shared__ __align__(32) __bf16 H1[2][8][32][16];   // 16 KB: h chunk
  __shared__ int toks_s[MTILE];
  __shared__ float gw_s[MTILE];

  const int tid = threadIdx.x;
  const int lane = tid & 31;
  const int wave = tid >> 5;

  if (tid < MTILE) {
    int s = base + tid;
    if (s < cnt) {
      toks_s[tid] = lists[e * NTOK + s];
      gw_s[tid] = gwbuf[e * NTOK + s];
    } else {
      toks_s[tid] = -1;
      gw_s[tid] = 0.f;
    }
  }
  __syncthreads();

  // Stage X tiles into WMMA A layout:
  // lane m (0-15): elems 0-15 -> K..K+15 ; lane m+16: K+16..K+31.
  for (int idx = tid; idx < 2 * 24 * 32 * 16; idx += 256) {
    int rt = idx / (24 * 512);
    int r2 = idx - rt * (24 * 512);
    int kb = r2 >> 9;
    int rem = r2 & 511;
    int ln = rem >> 4;
    int el = rem & 15;
    int m = ln & 15;
    int k = kb * 32 + el + ((ln & 16) ? 16 : 0);
    int tok = toks_s[rt * 16 + m];
    float v = (tok >= 0) ? hs[(size_t)tok * Hdim + k] : 0.f;
    Xa[rt][kb][ln][el] = (__bf16)v;
  }
  __syncthreads();

  const __bf16* w1se = w1s + (size_t)e * Hdim * Idim;  // [NB1][KB1][32][16]
  const __bf16* w2se = w2s + (size_t)e * Idim * Hdim;  // [NB2][KB2][32][16]
  const float* b1e = b1 + e * Idim;
  const float* b2e = b2 + e * Hdim;

  v8f oacc[2][6];
#pragma unroll
  for (int rt = 0; rt < 2; ++rt)
#pragma unroll
    for (int q = 0; q < 6; ++q)
      oacc[rt][q] = (v8f){0.f, 0.f, 0.f, 0.f, 0.f, 0.f, 0.f, 0.f};

  const int n = lane & 15;
  const int mofs = (lane & 16) ? 8 : 0;  // C/D: row = r + mofs

  for (int c = 0; c < 12; ++c) {
    // ---- phase 1: h chunk = gelu(X @ W1[:, c*256 .. +256) + b1)
    v8f acc[2][2];  // [j][row tile]
#pragma unroll
    for (int j = 0; j < 2; ++j)
#pragma unroll
      for (int rt = 0; rt < 2; ++rt)
        acc[j][rt] = (v8f){0.f, 0.f, 0.f, 0.f, 0.f, 0.f, 0.f, 0.f};
    for (int kb = 0; kb < 24; ++kb) {
      v16bf a0 = *(const v16bf*)&Xa[0][kb][lane][0];
      v16bf a1 = *(const v16bf*)&Xa[1][kb][lane][0];
#pragma unroll
      for (int j = 0; j < 2; ++j) {
        int nb = c * 16 + wave * 2 + j;  // column block of W1
        v16bf b = *(const v16bf*)(w1se + (((size_t)nb * KB1 + kb) * 32 + lane) * 16);
        acc[j][0] = __builtin_amdgcn_wmma_f32_16x16x32_bf16(
            false, a0, false, b, (short)0, acc[j][0], false, false);
        acc[j][1] = __builtin_amdgcn_wmma_f32_16x16x32_bf16(
            false, a1, false, b, (short)0, acc[j][1], false, false);
      }
    }
#pragma unroll
    for (int j = 0; j < 2; ++j) {
      int iblk = wave * 2 + j;
      float bias = b1e[c * 256 + iblk * 16 + n];
      int i_local = iblk * 16 + n;  // 0..255 within chunk
      int kb2 = i_local >> 5;
      int kl = i_local & 31;
      int el = kl & 15;
      int ln2b = (kl & 16) ? 16 : 0;
#pragma unroll
      for (int r = 0; r < 8; ++r) {
        int m = r + mofs;
        float x0 = acc[j][0][r] + bias;
        float x1 = acc[j][1][r] + bias;
        float g0 = 0.5f * x0 * (1.f + erff(x0 * 0.70710678118654752f));
        float g1 = 0.5f * x1 * (1.f + erff(x1 * 0.70710678118654752f));
        H1[0][kb2][m + ln2b][el] = (__bf16)g0;
        H1[1][kb2][m + ln2b][el] = (__bf16)g1;
      }
    }
    __syncthreads();
    // ---- phase 2: oacc += h_chunk @ W2[c*256 .., :]
#pragma unroll
    for (int kb2 = 0; kb2 < 8; ++kb2) {
      v16bf a0 = *(const v16bf*)&H1[0][kb2][lane][0];
      v16bf a1 = *(const v16bf*)&H1[1][kb2][lane][0];
#pragma unroll
      for (int q = 0; q < 6; ++q) {
        int nb2 = wave * 6 + q;  // column block of W2 (over H)
        v16bf b = *(const v16bf*)(
            w2se + (((size_t)nb2 * KB2 + c * 8 + kb2) * 32 + lane) * 16);
        oacc[0][q] = __builtin_amdgcn_wmma_f32_16x16x32_bf16(
            false, a0, false, b, (short)0, oacc[0][q], false, false);
        oacc[1][q] = __builtin_amdgcn_wmma_f32_16x16x32_bf16(
            false, a1, false, b, (short)0, oacc[1][q], false, false);
      }
    }
    __syncthreads();
  }

  // ---- epilogue: + b2, scale by gate, scatter-accumulate
#pragma unroll
  for (int q = 0; q < 6; ++q) {
    int h = (wave * 6 + q) * 16 + n;
    float bias = b2e[h];
#pragma unroll
    for (int rt = 0; rt < 2; ++rt) {
#pragma unroll
      for (int r = 0; r < 8; ++r) {
        int m = rt * 16 + r + mofs;
        int tok = toks_s[m];
        if (tok >= 0) {
          float val = oacc[rt][q][r] + bias;
          atomicAdd(&out[(size_t)tok * Hdim + h], gw_s[m] * val);
        }
      }
    }
  }
}

// ---------------------------------------------------------------------------
// Launch wiring
// ---------------------------------------------------------------------------
extern "C" void kernel_launch(void* const* d_in, const int* in_sizes, int n_in,
                              void* d_out, int out_size, void* d_ws,
                              size_t ws_size, hipStream_t stream) {
  const float* hs = (const float*)d_in[0];  // [B,S,H]
  const float* rw = (const float*)d_in[1];  // [H,E]
  const float* rb = (const float*)d_in[2];  // [E]
  const float* w1 = (const float*)d_in[3];  // [E,H,I]
  const float* b1 = (const float*)d_in[4];  // [E,I]
  const float* w2 = (const float*)d_in[5];  // [E,I,H]
  const float* b2 = (const float*)d_in[6];  // [E,H]

  float* out = (float*)d_out;
  const size_t BSH = (size_t)Bdim * Sdim * Hdim;      // 6291456
  const size_t BSE = (size_t)Bdim * Sdim * Edim;      // 65536
  float* final_out = out;                              // [B,S,H]
  float* aux_out = out + BSH;                          // scalar
  float* gating_out = out + BSH + 1;                   // [B,S,E]
  const size_t n_out_total = BSH + 1 + BSE;

  // workspace layout
  char* ws = (char*)d_ws;
  int* counts = (int*)ws;                              // [E]
  float* importance = (float*)(ws + 64);               // [E]
  int* lists = (int*)(ws + 128);                       // [E][NTOK]
  float* gwbuf = (float*)(ws + 128 + sizeof(int) * Edim * NTOK);
  size_t ofs = 128 + (sizeof(int) + sizeof(float)) * (size_t)Edim * NTOK;
  ofs = (ofs + 255) & ~(size_t)255;
  __bf16* w1s = (__bf16*)(ws + ofs);                   // [E*H*I] bf16 swizzled
  ofs += sizeof(__bf16) * (size_t)Edim * Hdim * Idim;
  __bf16* w2s = (__bf16*)(ws + ofs);                   // [E*I*H] bf16 swizzled

  zero_kernel<<<256, 256, 0, stream>>>(out, n_out_total, counts, importance);

  {
    int total = 2 * Edim * (Hdim / 32) * (Idim / 16) * 32;  // w1 tiles + w2 tiles
    prep_weights_kernel<<<total / 256, 256, 0, stream>>>(w1, w2, w1s, w2s);
  }

  router_kernel<<<NTOK / 256, 256, 0, stream>>>(hs, rw, rb, gating_out, lists,
                                                gwbuf, counts, importance);

  aux_kernel<<<1, 1, 0, stream>>>(importance, aux_out);

  dim3 grid(NTOK / MTILE, Edim);  // 256 tiles x 8 experts; empty tiles exit
  moe_ffn_kernel<<<grid, 256, 0, stream>>>(hs, w1s, b1, w2s, b2, lists, gwbuf,
                                           counts, final_out);
}
